// LSTM_23493471109987
// MI455X (gfx1250) — matrix operand order repaired and checked by reading, not compiled, not run on hardware
//
#include <hip/hip_runtime.h>

// ---------------- CDNA5 WMMA types ----------------
typedef __attribute__((ext_vector_type(16))) __bf16 v16bf;
typedef __attribute__((ext_vector_type(8)))  float  v8f;

union ABfrag { uint4 q[2]; v16bf v; };   // 32 bytes: 16 bf16 per lane

#define B_DIM 32
#define T_DIM 2048
#define D_DIM 512
#define H_DIM 512
#define FOURH 2048
#define M_DIM (T_DIM * B_DIM)   // 65536 rows of x_proj
#define NWG 16                  // workgroups in recurrent phase

__device__ __forceinline__ unsigned short f2bf(float f) {
    unsigned int u = __float_as_uint(f);
    unsigned int r = (u + 0x7FFFu + ((u >> 16) & 1u)) >> 16;   // RNE
    return (unsigned short)r;
}
__device__ __forceinline__ float bf2f(unsigned short h) {
    return __uint_as_float(((unsigned int)h) << 16);
}
__device__ __forceinline__ float sigmoidf_(float x) {
    return 1.0f / (1.0f + __expf(-x));
}
__device__ __forceinline__ float tanhf_(float x) {
    x = fminf(15.0f, fmaxf(-15.0f, x));
    float e = __expf(2.0f * x);
    return (e - 1.0f) / (e + 1.0f);
}

// ---------------- conversion kernels ----------------
// input [B,T,D] f32 -> xT [T*B, D] bf16 (row m = t*B+b)
__global__ __launch_bounds__(256) void cvt_input_kernel(
    const float* __restrict__ in, unsigned short* __restrict__ xT) {
    size_t i = (size_t)blockIdx.x * blockDim.x + threadIdx.x;
    if (i >= (size_t)B_DIM * T_DIM * D_DIM) return;
    int d = (int)(i % D_DIM);
    size_t bt = i / D_DIM;
    int t = (int)(bt % T_DIM);
    int b = (int)(bt / T_DIM);
    xT[((size_t)t * B_DIM + b) * D_DIM + d] = f2bf(in[i]);
}

__global__ __launch_bounds__(256) void cvt_f32_bf16_kernel(
    const float* __restrict__ src, unsigned short* __restrict__ dst, int n) {
    int i = blockIdx.x * blockDim.x + threadIdx.x;
    if (i < n) dst[i] = f2bf(src[i]);
}

// h0 -> hbuf[0] (bf16), counters[0]=NWG, counters[1..T]=0
__global__ __launch_bounds__(256) void init_kernel(
    const float* __restrict__ h0, unsigned short* __restrict__ hbuf,
    int* __restrict__ counters) {
    int i = blockIdx.x * blockDim.x + threadIdx.x;
    if (i < B_DIM * H_DIM) hbuf[i] = f2bf(h0[i]);
    if (i == 0) counters[0] = NWG;
    else if (i <= T_DIM) counters[i] = 0;
}

// ---------------- phase 1: x_proj GEMM (WMMA bf16) ----------------
// xp[m, g] = sum_d xT[m,d] * Wih[g,d] + b_ih[g] + b_hh[g]   (stored bf16)
// Workgroup tile 64M x 256N, 8 waves, wave tile 32M x 64N.
__global__ __launch_bounds__(256) void xproj_gemm_kernel(
    const unsigned short* __restrict__ xT,   // [M_DIM, D] bf16
    const unsigned short* __restrict__ Wih,  // [4H, D] bf16
    const float* __restrict__ b_ih, const float* __restrict__ b_hh,
    unsigned short* __restrict__ xp) {       // [M_DIM, 4H] bf16
    const int lane = threadIdx.x & 31;
    const int wave = threadIdx.x >> 5;
    const int lm = lane & 15;
    const int sel = lane >> 4;               // 0: lanes 0-15, 1: lanes 16-31
    const int m0 = blockIdx.x * 64 + (wave & 1) * 32;
    const int n0 = blockIdx.y * 256 + (wave >> 1) * 64;

    v8f acc[8];
#pragma unroll
    for (int i = 0; i < 8; ++i) acc[i] = (v8f){0,0,0,0,0,0,0,0};

    const size_t rowA0 = (size_t)(m0 + lm) * D_DIM;
    const size_t rowA1 = (size_t)(m0 + 16 + lm) * D_DIM;

    for (int k = 0; k < D_DIM; k += 32) {
        ABfrag a0, a1;
        a0.q[0] = *(const uint4*)(xT + rowA0 + k + sel * 8);
        a0.q[1] = *(const uint4*)(xT + rowA0 + k + 16 + sel * 8);
        a1.q[0] = *(const uint4*)(xT + rowA1 + k + sel * 8);
        a1.q[1] = *(const uint4*)(xT + rowA1 + k + 16 + sel * 8);
#pragma unroll
        for (int ns = 0; ns < 4; ++ns) {
            ABfrag bf;
            const size_t rowB = (size_t)(n0 + ns * 16 + lm) * D_DIM + k + sel * 8;
            bf.q[0] = *(const uint4*)(Wih + rowB);
            bf.q[1] = *(const uint4*)(Wih + rowB + 16);
            acc[2 * ns] = __builtin_amdgcn_wmma_f32_16x16x32_bf16(
                false, a0.v, false, bf.v, (short)0, acc[2 * ns], false, false);
            acc[2 * ns + 1] = __builtin_amdgcn_wmma_f32_16x16x32_bf16(
                false, a1.v, false, bf.v, (short)0, acc[2 * ns + 1], false, false);
        }
    }
#pragma unroll
    for (int ns = 0; ns < 4; ++ns) {
        int n = n0 + ns * 16 + lm;
        float bias = b_ih[n] + b_hh[n];
#pragma unroll
        for (int mt = 0; mt < 2; ++mt) {
            v8f c = acc[2 * ns + mt];
            int mbase = m0 + mt * 16 + sel * 8;
#pragma unroll
            for (int r = 0; r < 8; ++r)
                xp[(size_t)(mbase + r) * FOURH + n] = f2bf(c[r] + bias);
        }
    }
}

// ---------------- phase 2: persistent recurrence ----------------
// 16 WGs, WG g owns h columns [g*32, g*32+32) => gate columns q*H + g*32 + r.
// Lock-step over t with release/acquire counters; per-step GEMM via WMMA.
__global__ __launch_bounds__(256) void recurrent_kernel(
    const unsigned short* __restrict__ xp,    // [T*B, 4H] bf16
    const unsigned short* __restrict__ Whh,   // [4H, H] bf16
    unsigned short* __restrict__ hbuf,        // [(T+1)*B, H] bf16
    const float* __restrict__ c0,             // [B, H] f32
    float* __restrict__ out,                  // h [B,H] then c [B,H]
    int* __restrict__ counters) {
    __shared__ float gates[B_DIM][129];       // [batch][i|f|g|o x 32], padded
    __shared__ float cst[B_DIM][32];          // c state for this WG's columns

    const int g = blockIdx.x;
    const int tix = threadIdx.x;
    const int lane = tix & 31;
    const int wave = tix >> 5;
    const int lm = lane & 15;
    const int sel = lane >> 4;

    // init c slice
#pragma unroll
    for (int p = 0; p < 4; ++p) {
        int idx = tix + p * 256;
        int b = idx >> 5, r = idx & 31;
        cst[b][r] = c0[(size_t)b * H_DIM + g * 32 + r];
    }
    __syncthreads();

    // wave -> 16 gate columns: group q = wave/2, sub-offset (wave&1)*16
    const int gate_base = (wave >> 1) * H_DIM + g * 32 + (wave & 1) * 16;
    const int jbase = wave * 16;              // local column in gates[][]
    const int ncol = gate_base + lm;
    const size_t rowB_base = (size_t)ncol * H_DIM + sel * 8;

    for (int t = 1; t <= T_DIM; ++t) {
        if (tix == 0) {
            while (__hip_atomic_load(&counters[t - 1], __ATOMIC_ACQUIRE,
                                     __HIP_MEMORY_SCOPE_AGENT) < NWG) {
                __builtin_amdgcn_s_sleep(1);
            }
        }
        __syncthreads();

        const unsigned short* hprev = hbuf + (size_t)(t - 1) * B_DIM * H_DIM;
        v8f acc0 = (v8f){0,0,0,0,0,0,0,0};
        v8f acc1 = (v8f){0,0,0,0,0,0,0,0};
        const size_t rA0 = (size_t)lm * H_DIM;
        const size_t rA1 = (size_t)(16 + lm) * H_DIM;
        for (int k = 0; k < H_DIM; k += 32) {
            ABfrag a0, a1, bf;
            a0.q[0] = *(const uint4*)(hprev + rA0 + k + sel * 8);
            a0.q[1] = *(const uint4*)(hprev + rA0 + k + 16 + sel * 8);
            a1.q[0] = *(const uint4*)(hprev + rA1 + k + sel * 8);
            a1.q[1] = *(const uint4*)(hprev + rA1 + k + 16 + sel * 8);
            bf.q[0] = *(const uint4*)(Whh + rowB_base + k);
            bf.q[1] = *(const uint4*)(Whh + rowB_base + k + 16);
            acc0 = __builtin_amdgcn_wmma_f32_16x16x32_bf16(
                false, a0.v, false, bf.v, (short)0, acc0, false, false);
            acc1 = __builtin_amdgcn_wmma_f32_16x16x32_bf16(
                false, a1.v, false, bf.v, (short)0, acc1, false, false);
        }

        // add x_proj and stage gates into LDS
        const unsigned short* xpt = xp + (size_t)(t - 1) * B_DIM * FOURH;
        const int j = jbase + lm;
#pragma unroll
        for (int r = 0; r < 8; ++r) {
            int b0 = sel * 8 + r;
            gates[b0][j]      = acc0[r] + bf2f(xpt[(size_t)b0 * FOURH + ncol]);
            gates[16 + b0][j] = acc1[r] + bf2f(xpt[(size_t)(16 + b0) * FOURH + ncol]);
        }
        __syncthreads();

        // gate nonlinearities + state update (32x32 elements, 4 per thread)
        unsigned short* hnext = hbuf + (size_t)t * B_DIM * H_DIM;
#pragma unroll
        for (int p = 0; p < 4; ++p) {
            int idx = tix + p * 256;
            int b = idx >> 5, r = idx & 31;
            float iv = gates[b][r];
            float fv = gates[b][32 + r];
            float gv = gates[b][64 + r];
            float ov = gates[b][96 + r];
            float c  = cst[b][r];
            float cn = sigmoidf_(fv) * c + sigmoidf_(iv) * tanhf_(gv);
            float hn = sigmoidf_(ov) * tanhf_(cn);
            cst[b][r] = cn;
            hnext[(size_t)b * H_DIM + g * 32 + r] = f2bf(hn);
            if (t == T_DIM) {
                out[(size_t)b * H_DIM + g * 32 + r] = hn;
                out[(size_t)B_DIM * H_DIM + (size_t)b * H_DIM + g * 32 + r] = cn;
            }
        }
        __threadfence();           // make h stores visible device-wide
        __syncthreads();
        if (tix == 0) {
            __hip_atomic_fetch_add(&counters[t], 1, __ATOMIC_RELEASE,
                                   __HIP_MEMORY_SCOPE_AGENT);
        }
    }
}

// ---------------- host launch ----------------
extern "C" void kernel_launch(void* const* d_in, const int* in_sizes, int n_in,
                              void* d_out, int out_size, void* d_ws, size_t ws_size,
                              hipStream_t stream) {
    const float* input = (const float*)d_in[0];  // [B,T,D]
    const float* h0    = (const float*)d_in[1];  // [B,H]
    const float* c0    = (const float*)d_in[2];  // [B,H]
    const float* W_ih  = (const float*)d_in[3];  // [4H,D]
    const float* W_hh  = (const float*)d_in[4];  // [4H,H]
    const float* b_ih  = (const float*)d_in[5];
    const float* b_hh  = (const float*)d_in[6];
    float* out = (float*)d_out;

    // workspace carve-up (bf16 buffers), 256B aligned
    char* ws = (char*)d_ws;
    size_t off = 0;
    auto carve = [&](size_t bytes) {
        size_t r = off;
        off = (off + bytes + 255) & ~(size_t)255;
        return r;
    };
    unsigned short* xT   = (unsigned short*)(ws + carve((size_t)M_DIM * D_DIM * 2));
    unsigned short* Wihb = (unsigned short*)(ws + carve((size_t)FOURH * D_DIM * 2));
    unsigned short* Whhb = (unsigned short*)(ws + carve((size_t)FOURH * H_DIM * 2));
    unsigned short* xpb  = (unsigned short*)(ws + carve((size_t)M_DIM * FOURH * 2));
    unsigned short* hbuf = (unsigned short*)(ws + carve((size_t)(T_DIM + 1) * B_DIM * H_DIM * 2));
    int* counters        = (int*)(ws + carve((size_t)(T_DIM + 1) * sizeof(int)));
    (void)ws_size; (void)in_sizes; (void)n_in; (void)out_size;

    // 1. convert input (transpose b,t) and weights to bf16
    {
        size_t n = (size_t)B_DIM * T_DIM * D_DIM;
        cvt_input_kernel<<<dim3((unsigned)((n + 255) / 256)), dim3(256), 0, stream>>>(input, xT);
    }
    cvt_f32_bf16_kernel<<<dim3((FOURH * D_DIM + 255) / 256), dim3(256), 0, stream>>>(W_ih, Wihb, FOURH * D_DIM);
    cvt_f32_bf16_kernel<<<dim3((FOURH * H_DIM + 255) / 256), dim3(256), 0, stream>>>(W_hh, Whhb, FOURH * H_DIM);

    // 2. init h0 -> hbuf[0], counters
    init_kernel<<<dim3((B_DIM * H_DIM + 255) / 256), dim3(256), 0, stream>>>(h0, hbuf, counters);

    // 3. input projection GEMM (WMMA): grid 1024 (M) x 8 (N)
    xproj_gemm_kernel<<<dim3(M_DIM / 64, FOURH / 256), dim3(256), 0, stream>>>(
        xT, Wihb, b_ih, b_hh, xpb);

    // 4. persistent lock-step recurrence (WMMA per step)
    recurrent_kernel<<<dim3(NWG), dim3(256), 0, stream>>>(
        xpb, Whhb, hbuf, c0, out, counters);
}
